// ModalEncoder_54202487275605
// MI455X (gfx1250) — matrix operand order repaired and checked by reading, compiled, not run on hardware
//
#include <hip/hip_runtime.h>

typedef __attribute__((ext_vector_type(16))) __bf16 v16bf;
typedef __attribute__((ext_vector_type(8)))  __bf16 v8bf;
typedef __attribute__((ext_vector_type(8)))  float  v8f;

#define N_NODES 100000
#define N_EDGES 1600000
#define D_IN    384
#define D_HID   768
#define D_OUT   256
#define ROWS    32      // node rows per workgroup (100000 / 32 = 3125 exact)

// ---------------------------------------------------------------------------
// WMMA fragment loaders (CDNA5 v_wmma_f32_16x16x32_bf16 layouts, ISA 7.12.2)
// A (16x32 bf16, MxK): lane L -> row m=L%16; elems [0..7] = K k0+8*(L/16)..,
//                                     elems [8..15] = K k0+16+8*(L/16)..
// B (32x16 bf16, KxN): lane L -> col n=L%16; elems [0..7] = K k0+8*(L/16)..,
//                                     elems [8..15] = K k0+16+8*(L/16)..
// ---------------------------------------------------------------------------
__device__ __forceinline__ v16bf frag_a(const __bf16* base, int ld, int lane) {
    const int m = lane & 15, half = lane >> 4;
    const __bf16* p = base + m * ld + half * 8;
    v8bf lo = *(const v8bf*)(p);
    v8bf hi = *(const v8bf*)(p + 16);
    return __builtin_shufflevector(lo, hi, 0,1,2,3,4,5,6,7,8,9,10,11,12,13,14,15);
}

__device__ __forceinline__ v16bf frag_b(const __bf16* base, int ld, int lane) {
    const int n = lane & 15, half = lane >> 4;
    const __bf16* p = base + n * ld + half * 8;
    v8bf lo = *(const v8bf*)(p);
    v8bf hi = *(const v8bf*)(p + 16);
    return __builtin_shufflevector(lo, hi, 0,1,2,3,4,5,6,7,8,9,10,11,12,13,14,15);
}

#define WMMA_BF16(A, B, C) \
    __builtin_amdgcn_wmma_f32_16x16x32_bf16(false, (A), false, (B), (short)0, (C), false, false)

// Async DMA: 16 bytes global -> LDS (CDNA5 GLOBAL_LOAD_ASYNC_TO_LDS_B128,
// tracked by ASYNCcnt). lds_addr = wave-relative LDS byte address (low 32
// bits of the flat pointer per ISA §10.2 aperture mapping).
__device__ __forceinline__ void async_g2l_b128(unsigned lds_addr, const void* gaddr) {
    asm volatile("global_load_async_to_lds_b128 %0, %1, off"
                 :: "v"(lds_addr), "v"(gaddr) : "memory");
}
__device__ __forceinline__ void wait_asynccnt0() {
    asm volatile("s_wait_asynccnt 0x0" ::: "memory");
}

// ---------------------------------------------------------------------------
// 1) Weight transpose + fp32 -> bf16 convert: W[K][N] -> Wt[N][K]
// ---------------------------------------------------------------------------
__global__ __launch_bounds__(256) void transpose_bf16_kernel(
    const float* __restrict__ W, __bf16* __restrict__ Wt, int K, int N)
{
    int idx = blockIdx.x * 256 + threadIdx.x;
    if (idx < K * N) {
        int k = idx / N, n = idx - k * N;
        Wt[n * K + k] = (__bf16)W[k * N + n];
    }
}

// ---------------------------------------------------------------------------
// 2) Zero scratch (accum + cnt)
// ---------------------------------------------------------------------------
__global__ __launch_bounds__(256) void zero_kernel(float* __restrict__ p, int n)
{
    int i = blockIdx.x * 256 + threadIdx.x;
    if (i < n) p[i] = 0.0f;
}

// ---------------------------------------------------------------------------
// 3) Fused MLP encoder: h = GELU(LN(x@W1 + b1)) @ W2 + b2
//    One WG = 32 node rows, 8 waves, 72 KB LDS (x tile 24K + hidden 48K,
//    hidden kept bf16 end-to-end; LN stats in fp32).
//    Emits h in fp32 (for scatter/residual) and bf16 (for sage WMMA A).
// ---------------------------------------------------------------------------
__global__ __launch_bounds__(256) void encoder_kernel(
    const float* __restrict__ x,
    const __bf16* __restrict__ Wt1, const float* __restrict__ b1,
    const float* __restrict__ g1,  const float* __restrict__ be1,
    const __bf16* __restrict__ Wt2, const float* __restrict__ b2,
    float* __restrict__ h, __bf16* __restrict__ hbg)
{
    __shared__ __align__(16) __bf16 xs[ROWS * D_IN];   // 24 KB x tile (bf16)
    __shared__ __align__(16) __bf16 hs[ROWS * D_HID];  // 48 KB hidden (bf16)

    const int tid  = threadIdx.x;
    const int row0 = blockIdx.x * ROWS;

    // Stage x tile as bf16 into LDS
    for (int i = tid; i < ROWS * D_IN; i += 256) {
        int r = i / D_IN, k = i - r * D_IN;
        xs[r * D_IN + k] = (__bf16)x[(row0 + r) * D_IN + k];
    }
    __syncthreads();

    const int wave = tid >> 5, lane = tid & 31;
    const int m = lane & 15, half = lane >> 4;

    // GEMM1: [32 x 384] @ [384 x 768]; 2 M-tiles x 48 N-tiles over 8 waves
    for (int t = wave; t < 2 * (D_HID / 16); t += 8) {
        const int mt = t & 1, nt = t >> 1;
        if (t + 8 < 2 * (D_HID / 16))
            __builtin_prefetch(Wt1 + (nt + 4) * 16 * D_IN, 0, 1);  // global_prefetch_b8
        v8f acc = {};
        for (int k0 = 0; k0 < D_IN; k0 += 32) {
            v16bf a = frag_a(xs + mt * 16 * D_IN + k0, D_IN, lane);   // ds_load_b128 x2
            v16bf b = frag_b(Wt1 + nt * 16 * D_IN + k0, D_IN, lane);  // global_load_b128 x2
            acc = WMMA_BF16(a, b, acc);
        }
        float bias = b1[nt * 16 + m];
        // C layout: elem r of lane L -> (M = r + 8*(L/16), N = L%16)
        for (int r = 0; r < 8; ++r)
            hs[(mt * 16 + r + 8 * half) * D_HID + nt * 16 + m] = (__bf16)(acc[r] + bias);
    }
    __syncthreads();

    // LayerNorm + exact GELU in place (8 lanes per row, shfl_xor reduction)
    {
        int r = tid >> 3, sub = tid & 7;
        float s = 0.f, s2 = 0.f;
        for (int k = sub; k < D_HID; k += 8) {
            float v = (float)hs[r * D_HID + k];
            s += v; s2 += v * v;
        }
        for (int off = 1; off < 8; off <<= 1) {
            s  += __shfl_xor(s,  off, 32);
            s2 += __shfl_xor(s2, off, 32);
        }
        float mu  = s  * (1.0f / D_HID);
        float var = s2 * (1.0f / D_HID) - mu * mu;
        float inv = rsqrtf(var + 1e-5f);
        for (int k = sub; k < D_HID; k += 8) {
            float v = ((float)hs[r * D_HID + k] - mu) * inv * g1[k] + be1[k];
            v = 0.5f * v * (1.0f + erff(v * 0.70710678118654752f));   // exact GELU
            hs[r * D_HID + k] = (__bf16)v;
        }
    }
    __syncthreads();

    // GEMM2: [32 x 768] @ [768 x 256] -> h (fp32) + hbg (bf16)
    for (int t = wave; t < 2 * (D_OUT / 16); t += 8) {
        const int mt = t & 1, nt = t >> 1;
        v8f acc = {};
        for (int k0 = 0; k0 < D_HID; k0 += 32) {
            v16bf a = frag_a(hs + mt * 16 * D_HID + k0, D_HID, lane);
            v16bf b = frag_b(Wt2 + nt * 16 * D_HID + k0, D_HID, lane);
            acc = WMMA_BF16(a, b, acc);
        }
        float bias = b2[nt * 16 + m];
        for (int r = 0; r < 8; ++r) {
            float v = acc[r] + bias;
            int row = row0 + mt * 16 + r + 8 * half;
            h[row * D_OUT + nt * 16 + m]   = v;
            hbg[row * D_OUT + nt * 16 + m] = (__bf16)v;
        }
    }
}

// ---------------------------------------------------------------------------
// 4) Edge scatter: accum[dst] += h[src]; cnt[dst] += 1.  64 lanes per edge,
//    edge indices staged once per block in LDS; float4 gather + 4
//    global_atomic_add_f32 per lane. Memory-bound stage (accum fits in L2).
// ---------------------------------------------------------------------------
__global__ __launch_bounds__(256) void scatter_kernel(
    const float* __restrict__ h, const long long* __restrict__ ei,
    float* __restrict__ accum, float* __restrict__ cnt)
{
    __shared__ long long se[8];      // 4 src + 4 dst ids for this block's edges
    const int e0 = (blockIdx.x * 256) >> 6;   // first edge of block
    if (threadIdx.x < 8) {
        int j = threadIdx.x;
        se[j] = (j < 4) ? ei[e0 + j] : ei[N_EDGES + e0 + (j - 4)];
    }
    __syncthreads();
    const int le = threadIdx.x >> 6;   // local edge 0..3
    const int c  = threadIdx.x & 63;   // 64 chunks of 4 floats
    const int src = (int)se[le];
    const int dst = (int)se[4 + le];
    const float4 v = *(const float4*)(h + src * D_OUT + c * 4);
    float* a = accum + dst * D_OUT + c * 4;
    atomicAdd(a + 0, v.x);
    atomicAdd(a + 1, v.y);
    atomicAdd(a + 2, v.z);
    atomicAdd(a + 3, v.w);
    if (c == 0) atomicAdd(cnt + dst, 1.0f);
}

// ---------------------------------------------------------------------------
// 5) SAGE combine + residual LN:
//    out = LN(h + (accum/max(cnt,1)) @ Wl + bl + h @ Wr)
//    One WG = 32 rows, 8 waves, 64 KB LDS. h tile DMA'd in as bf16 via
//    GLOBAL_LOAD_ASYNC_TO_LDS_B128 (ASYNCcnt path).
// ---------------------------------------------------------------------------
__global__ __launch_bounds__(256) void sage_kernel(
    const float* __restrict__ h, const __bf16* __restrict__ hbg,
    const float* __restrict__ accum, const float* __restrict__ cnt,
    const __bf16* __restrict__ Wtl, const float* __restrict__ bl,
    const __bf16* __restrict__ Wtr,
    const float* __restrict__ g2, const float* __restrict__ be2,
    float* __restrict__ out)
{
    __shared__ __align__(16) __bf16 mb[ROWS * D_OUT];  // mean, bf16 (16 KB)
    __shared__ __align__(16) __bf16 hb[ROWS * D_OUT];  // h, bf16   (16 KB)
    __shared__ __align__(16) float  sg[ROWS * D_OUT];  // sage fp32 (32 KB)

    const int tid  = threadIdx.x;
    const int row0 = blockIdx.x * ROWS;

    // Async DMA the contiguous 16 KB bf16 h tile straight into LDS.
    {
        const char* g = (const char*)(hbg + (size_t)row0 * D_OUT);
        unsigned lbase = (unsigned)(uintptr_t)(&hb[0]);
        for (int i = 0; i < 4; ++i)
            async_g2l_b128(lbase + tid * 16 + i * 4096, g + tid * 16 + i * 4096);
    }
    // Meanwhile compute mean -> bf16 with VALU (needs the divide anyway)
    for (int i = tid; i < ROWS * D_OUT; i += 256) {
        int r = i >> 8, k = i & (D_OUT - 1);
        float c = cnt[row0 + r];
        c = (c > 1.0f) ? c : 1.0f;
        mb[i] = (__bf16)(accum[(row0 + r) * D_OUT + k] / c);
    }
    wait_asynccnt0();
    __syncthreads();

    const int wave = tid >> 5, lane = tid & 31;
    const int m = lane & 15, half = lane >> 4;

    // sage = mean @ Wl + h @ Wr + bl  (two K=256 passes into one accumulator)
    for (int t = wave; t < 2 * (D_OUT / 16); t += 8) {
        const int mt = t & 1, nt = t >> 1;
        v8f acc = {};
        for (int k0 = 0; k0 < D_OUT; k0 += 32) {
            v16bf a = frag_a(mb + mt * 16 * D_OUT + k0, D_OUT, lane);
            v16bf b = frag_b(Wtl + nt * 16 * D_OUT + k0, D_OUT, lane);
            acc = WMMA_BF16(a, b, acc);
        }
        for (int k0 = 0; k0 < D_OUT; k0 += 32) {
            v16bf a = frag_a(hb + mt * 16 * D_OUT + k0, D_OUT, lane);
            v16bf b = frag_b(Wtr + nt * 16 * D_OUT + k0, D_OUT, lane);
            acc = WMMA_BF16(a, b, acc);
        }
        float bias = bl[nt * 16 + m];
        for (int r = 0; r < 8; ++r)
            sg[(mt * 16 + r + 8 * half) * D_OUT + nt * 16 + m] = acc[r] + bias;
    }
    __syncthreads();

    // residual + LayerNorm (h read in fp32 for the residual)
    {
        int r = tid >> 3, sub = tid & 7;
        float s = 0.f, s2 = 0.f;
        for (int k = sub; k < D_OUT; k += 8) {
            float v = h[(row0 + r) * D_OUT + k] + sg[r * D_OUT + k];
            sg[r * D_OUT + k] = v;
            s += v; s2 += v * v;
        }
        for (int off = 1; off < 8; off <<= 1) {
            s  += __shfl_xor(s,  off, 32);
            s2 += __shfl_xor(s2, off, 32);
        }
        float mu  = s  * (1.0f / D_OUT);
        float var = s2 * (1.0f / D_OUT) - mu * mu;
        float inv = rsqrtf(var + 1e-5f);
        for (int k = sub; k < D_OUT; k += 8)
            out[(row0 + r) * D_OUT + k] = (sg[r * D_OUT + k] - mu) * inv * g2[k] + be2[k];
    }
}

// ---------------------------------------------------------------------------
// Host-side launch. Workspace layout (bytes, all 16B-aligned):
//   h (fp32 N*256) | accum (fp32 N*256) | cnt (fp32 N) | hbg (bf16 N*256)
//   | Wt1 | Wt2 | Wtl | Wtr  (bf16, transposed)           total ~258 MB
// ---------------------------------------------------------------------------
extern "C" void kernel_launch(void* const* d_in, const int* in_sizes, int n_in,
                              void* d_out, int out_size, void* d_ws, size_t ws_size,
                              hipStream_t stream)
{
    const float*     x   = (const float*)d_in[0];
    const long long* ei  = (const long long*)d_in[1];
    const float*     W1  = (const float*)d_in[2];
    const float*     b1  = (const float*)d_in[3];
    const float*     g1  = (const float*)d_in[4];
    const float*     be1 = (const float*)d_in[5];
    const float*     W2  = (const float*)d_in[6];
    const float*     b2  = (const float*)d_in[7];
    const float*     Wl  = (const float*)d_in[8];
    const float*     bl  = (const float*)d_in[9];
    const float*     Wr  = (const float*)d_in[10];
    const float*     g2  = (const float*)d_in[11];
    const float*     be2 = (const float*)d_in[12];
    float*           out = (float*)d_out;

    char* ws = (char*)d_ws;
    size_t off = 0;
    float*  h     = (float*)(ws + off);  off += (size_t)N_NODES * D_OUT * 4;
    float*  accum = (float*)(ws + off);  off += (size_t)N_NODES * D_OUT * 4;
    float*  cnt   = (float*)(ws + off);  off += (size_t)N_NODES * 4;
    __bf16* hbg   = (__bf16*)(ws + off); off += (size_t)N_NODES * D_OUT * 2;
    __bf16* Wt1   = (__bf16*)(ws + off); off += (size_t)D_HID * D_IN  * 2;
    __bf16* Wt2   = (__bf16*)(ws + off); off += (size_t)D_OUT * D_HID * 2;
    __bf16* Wtl   = (__bf16*)(ws + off); off += (size_t)D_OUT * D_OUT * 2;
    __bf16* Wtr   = (__bf16*)(ws + off); off += (size_t)D_OUT * D_OUT * 2;

    // 1) weights -> transposed bf16
    transpose_bf16_kernel<<<(D_IN * D_HID + 255) / 256, 256, 0, stream>>>(W1, Wt1, D_IN, D_HID);
    transpose_bf16_kernel<<<(D_HID * D_OUT + 255) / 256, 256, 0, stream>>>(W2, Wt2, D_HID, D_OUT);
    transpose_bf16_kernel<<<(D_OUT * D_OUT + 255) / 256, 256, 0, stream>>>(Wl, Wtl, D_OUT, D_OUT);
    transpose_bf16_kernel<<<(D_OUT * D_OUT + 255) / 256, 256, 0, stream>>>(Wr, Wtr, D_OUT, D_OUT);

    // 2) zero accum + cnt (contiguous)
    const int nzero = N_NODES * D_OUT + N_NODES;
    zero_kernel<<<(nzero + 255) / 256, 256, 0, stream>>>(accum, nzero);

    // 3) fused encoder -> h, hbg
    encoder_kernel<<<N_NODES / ROWS, 256, 0, stream>>>(x, Wt1, b1, g1, be1, Wt2, b2, h, hbg);

    // 4) edge scatter-add
    scatter_kernel<<<(N_EDGES * 64) / 256, 256, 0, stream>>>(h, ei, accum, cnt);

    // 5) SAGE combine + residual LN -> out
    sage_kernel<<<N_NODES / ROWS, 256, 0, stream>>>(h, hbg, accum, cnt, Wtl, bl, Wtr, g2, be2, out);
}
